// TransformerBackbone_38328288149703
// MI455X (gfx1250) — compile-verified, matrix operand order checked by reading, not run on hardware
//
#include <hip/hip_runtime.h>
#include <hip/hip_bf16.h>

typedef __attribute__((ext_vector_type(16))) _Float16 v16h;
typedef __attribute__((ext_vector_type(8)))  _Float16 v8h;
typedef __attribute__((ext_vector_type(8)))  float    v8f;

// ---------------- problem constants ----------------
constexpr int kB   = 2048;
constexpr int kD   = 1024;
constexpr int kH   = 8;
constexpr int kF   = 4096;
constexpr int kL   = 2;
constexpr int kRM  = 256;   // rel-pos clip M
constexpr int kMEM = 16;
constexpr int kACT = 32;
constexpr int kSTO = 1024;
constexpr int kHID = 1024;
constexpr int kS   = kMEM + 1;       // 17
constexpr int kHD  = kD / kH;        // 128
constexpr int kKIN = kSTO + kACT;    // 1056
constexpr int kBS  = kB * kS;        // 34816
constexpr float kSCALE = 0.08838834764831845f;   // 128^-0.5
constexpr float kEPS   = 1e-4f;

// ---------------- GEMM tile config ----------------
#define TILE_M 128
#define TILE_N 256
#define TILE_K 32
#define LDA_P (TILE_K + 8)   // A staged [TILE_M][LDA_P]   (K-major rows)
#define LDB_P (TILE_K + 8)   // B^T staged [TILE_N][LDB_P] (K-major rows)

union Frag16 {
  v16h v;
  v8h  h[2];
};

// ---- CDNA5 async global->LDS copy (ASYNCcnt) ----
__device__ __forceinline__ void async_b128(const _Float16* g, _Float16* l) {
  uint32_t lofs = (uint32_t)(uintptr_t)l;      // LDS byte address (low 32 bits of flat)
  asm volatile("global_load_async_to_lds_b128 %0, %1, off"
               :: "v"(lofs), "v"((const void*)g) : "memory");
}
__device__ __forceinline__ void wait_async() {
  asm volatile("s_wait_asynccnt 0" ::: "memory");
}

// ---------------- transposing f32 -> f16 weight convert --------------------
// src [batch][K][N] f32  ->  dst [batch][N][K] f16 (32x32 LDS tiles)
__global__ __launch_bounds__(256) void convert_transpose_kernel(
    const float* __restrict__ src, _Float16* __restrict__ dst, int K, int N) {
  __shared__ _Float16 t[32][33];
  size_t batch_off = (size_t)blockIdx.z * K * N;
  int k0 = blockIdx.x * 32, n0 = blockIdx.y * 32;
  int r = threadIdx.x >> 5, c = threadIdx.x & 31;   // 8 rows x 32 cols per pass
#pragma unroll
  for (int p = 0; p < 4; ++p)
    t[r + 8 * p][c] = (_Float16)src[batch_off + (size_t)(k0 + r + 8 * p) * N + n0 + c];
  __syncthreads();
#pragma unroll
  for (int p = 0; p < 4; ++p)
    dst[batch_off + (size_t)(n0 + r + 8 * p) * K + k0 + c] = t[c][r + 8 * p];
}

// ---------------- token input build (stoch || norm(action)) in f16 ---------
__global__ __launch_bounds__(256) void build_tok_in_kernel(
    const float* __restrict__ stoch, const float* __restrict__ action,
    _Float16* __restrict__ out) {
  int i = blockIdx.x * 256 + threadIdx.x;
  if (i >= kB * kKIN) return;
  int b = i / kKIN, c = i - b * kKIN;
  float v;
  if (c < kSTO) {
    v = stoch[(size_t)b * kSTO + c];
  } else {
    float a = action[(size_t)b * kACT + (c - kSTO)];
    v = a / fmaxf(fabsf(a), 1.0f);
  }
  out[i] = (_Float16)v;
}

// ---------------- seq = concat(memory, tok) ----------------
__global__ __launch_bounds__(256) void build_seq_kernel(
    const float* __restrict__ memory, const float* __restrict__ tok,
    float* __restrict__ seq) {
  int i = blockIdx.x * 256 + threadIdx.x;
  if (i >= kBS * kD) return;
  int d = i & (kD - 1);
  int bs = i >> 10;               // kD == 1024
  int s = bs % kS, b = bs / kS;
  float v;
  if (s < kMEM) v = memory[((size_t)b * kMEM + s) * kD + d];
  else          v = tok[(size_t)b * kD + d];
  seq[i] = v;
}

// ---------------- generic f16 WMMA GEMM with fused epilogue ----------------
// C[M,N] = A[M,K] @ B[K,N];  A row-major [M][K], B passed TRANSPOSED [N][K].
// Block tile 128x256, 8 waves (2x4), each wave 64x64 = 4x4 WMMA tiles
// (16 WMMAs per 16 ds_load_b128 -> 1.0 LDS reads per WMMA, 4x frag reuse).
// Double-buffered LDS fed by global_load_async_to_lds_b128 (ASYNCcnt).
// epilogue: x = acc + bias[col]; act(0=none,1=silu,2=gelu); x += resid; store.
__global__ __launch_bounds__(256) void gemm_f16_wmma_kernel(
    const _Float16* __restrict__ A, const _Float16* __restrict__ BT,
    const float* __restrict__ bias, const float* __restrict__ resid,
    float* __restrict__ outF, _Float16* __restrict__ outH,
    int Mdim, int Ndim, int Kdim, int act) {
  __shared__ _Float16 As[2][TILE_M * LDA_P];
  __shared__ _Float16 Bs[2][TILE_N * LDB_P];

  const int tid  = threadIdx.x;
  const int lane = tid & 31;
  const int w    = tid >> 5;           // 8 waves
  const int wr   = w >> 2;             // 0..1 : 64-row band
  const int wc   = w & 3;              // 0..3 : 64-col band
  const int half = lane >> 4;
  const int lm   = lane & 15;
  const int m0   = blockIdx.y * TILE_M;
  const int n0   = blockIdx.x * TILE_N;

  const v8f vzero = {0.f, 0.f, 0.f, 0.f, 0.f, 0.f, 0.f, 0.f};
  v8f acc[4][4];
#pragma unroll
  for (int tm = 0; tm < 4; ++tm)
#pragma unroll
    for (int tn = 0; tn < 4; ++tn) acc[tm][tn] = vzero;

  // stage one 128x32 A slab + 256x32 B^T slab into LDS buffer `buf`
  auto stage = [&](int buf, int k0) {
#pragma unroll
    for (int it = 0; it < 2; ++it) {
      int chunk = tid + it * 256;          // 512 chunks of 8 halves
      int r = chunk >> 2, c = (chunk & 3) << 3;
      async_b128(A + (size_t)(m0 + r) * Kdim + k0 + c, &As[buf][r * LDA_P + c]);
    }
#pragma unroll
    for (int it = 0; it < 4; ++it) {
      int chunk = tid + it * 256;          // 1024 chunks of 8 halves
      int r = chunk >> 2, c = (chunk & 3) << 3;
      async_b128(BT + (size_t)(n0 + r) * Kdim + k0 + c, &Bs[buf][r * LDB_P + c]);
    }
    // warm L2 for the slab after this one (global_prefetch_b8)
    if (k0 + TILE_K < Kdim) {
      __builtin_prefetch(A  + (size_t)(m0 + (tid >> 1)) * Kdim + k0 + TILE_K, 0, 1);
      __builtin_prefetch(BT + (size_t)(n0 + tid) * Kdim + k0 + TILE_K, 0, 1);
    }
  };

  stage(0, 0);
  int ib = 0;
  for (int k0 = 0; k0 < Kdim; k0 += TILE_K) {
    wait_async();          // this wave's async loads for tile `ib` complete
    __syncthreads();       // all waves' loads landed; previous reads finished
    if (k0 + TILE_K < Kdim) stage(ib ^ 1, k0 + TILE_K);

    // ---- fragments: two contiguous 16B LDS reads each ----
    Frag16 afrag[4], bfrag[4];
#pragma unroll
    for (int t = 0; t < 4; ++t) {
      const _Float16* ap = &As[ib][(wr * 64 + t * 16 + lm) * LDA_P + (half << 3)];
      afrag[t].h[0] = *(const v8h*)(ap);
      afrag[t].h[1] = *(const v8h*)(ap + 16);
    }
#pragma unroll
    for (int t = 0; t < 4; ++t) {
      const _Float16* bp = &Bs[ib][(wc * 64 + t * 16 + lm) * LDB_P + (half << 3)];
      bfrag[t].h[0] = *(const v8h*)(bp);
      bfrag[t].h[1] = *(const v8h*)(bp + 16);
    }
    // ---- 4x4 WMMA tiles per wave ----
#pragma unroll
    for (int tm = 0; tm < 4; ++tm)
#pragma unroll
      for (int tn = 0; tn < 4; ++tn)
        acc[tm][tn] = __builtin_amdgcn_wmma_f32_16x16x32_f16(
            false, afrag[tm].v, false, bfrag[tn].v, (short)0, acc[tm][tn],
            false, false);
    ib ^= 1;
  }

  // ---- epilogue: C/D layout: row = vgpr + 8*half, col = lane&15 ----
#pragma unroll
  for (int tm = 0; tm < 4; ++tm) {
#pragma unroll
    for (int tn = 0; tn < 4; ++tn) {
#pragma unroll
      for (int vv = 0; vv < 8; ++vv) {
        int row = m0 + wr * 64 + tm * 16 + vv + (half << 3);
        int col = n0 + wc * 64 + tn * 16 + lm;
        if (row < Mdim) {
          float x = acc[tm][tn][vv];
          if (bias) x += bias[col];
          if (act == 1) x = x / (1.0f + __expf(-x));                        // silu
          else if (act == 2) x = 0.5f * x * (1.0f + erff(x * 0.70710678f)); // gelu (exact)
          size_t idx = (size_t)row * Ndim + col;
          if (resid) x += resid[idx];
          if (outF) outF[idx] = x;
          if (outH) outH[idx] = (_Float16)x;
        }
      }
    }
  }
}

// ---------------- row-wise RMSNorm (f32 in -> f16/f32 out, optional silu) ---
__global__ __launch_bounds__(256) void rmsnorm_kernel(
    const float* __restrict__ in, const float* __restrict__ wgt,
    _Float16* __restrict__ outH, float* __restrict__ outF, int C, int act) {
  int row = blockIdx.x;
  const float* x = in + (size_t)row * C;
  int tid = threadIdx.x;
  float ss = 0.f;
  for (int c = tid; c < C; c += 256) { float v = x[c]; ss += v * v; }
#pragma unroll
  for (int off = 16; off; off >>= 1) ss += __shfl_down(ss, off, 32);
  __shared__ float red[8];
  __shared__ float sinv;
  if ((tid & 31) == 0) red[tid >> 5] = ss;
  __syncthreads();
  if (tid == 0) {
    float t = 0.f;
#pragma unroll
    for (int i = 0; i < 8; ++i) t += red[i];
    sinv = rsqrtf(t / (float)C + kEPS);
  }
  __syncthreads();
  float rn = sinv;
  for (int c = tid; c < C; c += 256) {
    float v = x[c] * rn * wgt[c];
    if (act == 1) v = v / (1.0f + __expf(-v));
    size_t idx = (size_t)row * C + c;
    if (outH) outH[idx] = (_Float16)v;
    if (outF) outF[idx] = v;
  }
}

// ---------------- attention: one workgroup per (b,h), S=17, HD=128 ---------
__global__ __launch_bounds__(256) void attention_kernel(
    const _Float16* __restrict__ qf, const _Float16* __restrict__ kf,
    const _Float16* __restrict__ vf, const float* __restrict__ rel,
    _Float16* __restrict__ att) {
  int bh = blockIdx.x;
  int b = bh >> 3, h = bh & 7;
  __shared__ float qs[kS][kHD + 1];
  __shared__ float ks[kS][kHD + 1];
  __shared__ float vs[kS][kHD + 1];
  __shared__ float sc[kS][kS + 1];
  int tid = threadIdx.x;
  for (int i = tid; i < kS * kHD; i += 256) {
    int s = i >> 7, d = i & (kHD - 1);
    size_t base = ((size_t)(b * kS + s)) * kD + h * kHD + d;
    qs[s][d] = (float)qf[base];
    ks[s][d] = (float)kf[base];
    vs[s][d] = (float)vf[base];
  }
  __syncthreads();
  for (int idx = tid; idx < kS * kS; idx += 256) {
    int i = idx / kS, j = idx - i * kS;
    float dot = 0.f;
#pragma unroll 8
    for (int d = 0; d < kHD; ++d) dot += qs[i][d] * ks[j][d];
    sc[i][j] = dot * kSCALE + rel[(size_t)(i - j + kRM) * kH + h];
  }
  __syncthreads();
  if (tid < kS) {
    int i = tid;
    float mx = -1e30f;
#pragma unroll
    for (int j = 0; j < kS; ++j) mx = fmaxf(mx, sc[i][j]);
    float sum = 0.f;
#pragma unroll
    for (int j = 0; j < kS; ++j) { float e = __expf(sc[i][j] - mx); sc[i][j] = e; sum += e; }
    float inv = 1.0f / sum;
#pragma unroll
    for (int j = 0; j < kS; ++j) sc[i][j] *= inv;
  }
  __syncthreads();
  for (int idx = tid; idx < kS * kHD; idx += 256) {
    int i = idx >> 7, d = idx & (kHD - 1);
    float o = 0.f;
#pragma unroll
    for (int j = 0; j < kS; ++j) o += sc[i][j] * vs[j][d];
    att[((size_t)(b * kS + i)) * kD + h * kHD + d] = (_Float16)o;
  }
}

// ---------------- final: new_deter = rmsnorm(seq[:, -1] + deter, fnw) ------
__global__ __launch_bounds__(256) void final_deter_kernel(
    const float* __restrict__ seq, const float* __restrict__ deter,
    const float* __restrict__ fnw, float* __restrict__ out) {
  int b = blockIdx.x;
  int tid = threadIdx.x;
  const float* srow = seq + ((size_t)b * kS + (kS - 1)) * kD;
  const float* drow = deter + (size_t)b * kD;
  float ss = 0.f;
  for (int c = tid; c < kD; c += 256) { float v = srow[c] + drow[c]; ss += v * v; }
#pragma unroll
  for (int off = 16; off; off >>= 1) ss += __shfl_down(ss, off, 32);
  __shared__ float red[8];
  __shared__ float sinv;
  if ((tid & 31) == 0) red[tid >> 5] = ss;
  __syncthreads();
  if (tid == 0) {
    float t = 0.f;
#pragma unroll
    for (int i = 0; i < 8; ++i) t += red[i];
    sinv = rsqrtf(t / (float)kD + kEPS);
  }
  __syncthreads();
  float rn = sinv;
  for (int c = tid; c < kD; c += 256)
    out[(size_t)b * kD + c] = (srow[c] + drow[c]) * rn * fnw[c];
}

// ---------------- new_mem = seq[:, 1:, :] ----------------
__global__ __launch_bounds__(256) void copy_mem_kernel(
    const float* __restrict__ seq, float* __restrict__ out) {
  int i = blockIdx.x * 256 + threadIdx.x;
  if (i >= kB * kMEM * kD) return;
  int d = i & (kD - 1);
  int bm = i >> 10;
  int m = bm % kMEM, b = bm / kMEM;
  out[i] = seq[((size_t)b * kS + m + 1) * kD + d];
}

// =====================================================================
extern "C" void kernel_launch(void* const* d_in, const int* in_sizes, int n_in,
                              void* d_out, int out_size, void* d_ws, size_t ws_size,
                              hipStream_t stream) {
  const float* stoch   = (const float*)d_in[0];
  const float* deter   = (const float*)d_in[1];
  const float* action  = (const float*)d_in[2];
  const float* memory  = (const float*)d_in[3];
  const float* inp_w1  = (const float*)d_in[4];
  const float* inp_b1  = (const float*)d_in[5];
  const float* inp_nw  = (const float*)d_in[6];
  const float* inp_w2  = (const float*)d_in[7];
  const float* inp_b2  = (const float*)d_in[8];
  const float* Wq      = (const float*)d_in[9];
  const float* bq      = (const float*)d_in[10];
  const float* Wk      = (const float*)d_in[11];
  const float* bk      = (const float*)d_in[12];
  const float* Wv      = (const float*)d_in[13];
  const float* bv      = (const float*)d_in[14];
  const float* Wo      = (const float*)d_in[15];
  const float* bo      = (const float*)d_in[16];
  const float* rel_emb = (const float*)d_in[17];
  const float* n1w     = (const float*)d_in[18];
  const float* n2w     = (const float*)d_in[19];
  const float* ffw1    = (const float*)d_in[20];
  const float* ffb1    = (const float*)d_in[21];
  const float* ffw2    = (const float*)d_in[22];
  const float* ffb2    = (const float*)d_in[23];
  const float* fnw     = (const float*)d_in[24];
  float* out = (float*)d_out;

  // ---- workspace carve-out ----
  char* ws = (char*)d_ws;
  size_t off = 0;
  auto carve = [&](size_t bytes) -> void* {
    void* p = ws + off;
    off = (off + bytes + 255) & ~(size_t)255;
    return p;
  };
  _Float16* w_inp1 = (_Float16*)carve((size_t)kKIN * kHID * 2);   // [HID][KIN]
  _Float16* w_inp2 = (_Float16*)carve((size_t)kHID * kD * 2);     // [D][HID]
  _Float16* w_q    = (_Float16*)carve((size_t)kL * kD * kD * 2);  // [l][D][D] (T)
  _Float16* w_k    = (_Float16*)carve((size_t)kL * kD * kD * 2);
  _Float16* w_v    = (_Float16*)carve((size_t)kL * kD * kD * 2);
  _Float16* w_o    = (_Float16*)carve((size_t)kL * kD * kD * 2);
  _Float16* w_f1   = (_Float16*)carve((size_t)kL * kD * kF * 2);  // [l][F][D] (T)
  _Float16* w_f2   = (_Float16*)carve((size_t)kL * kF * kD * 2);  // [l][D][F] (T)
  _Float16* tok_in = (_Float16*)carve((size_t)kB * kKIN * 2);
  float*    h1     = (float*)   carve((size_t)kB * kHID * 4);
  _Float16* h1n    = (_Float16*)carve((size_t)kB * kHID * 2);
  float*    tok    = (float*)   carve((size_t)kB * kD * 4);
  float*    seq    = (float*)   carve((size_t)kBS * kD * 4);
  _Float16* xn     = (_Float16*)carve((size_t)kBS * kD * 2);
  _Float16* qf     = (_Float16*)carve((size_t)kBS * kD * 2);
  _Float16* kfp    = (_Float16*)carve((size_t)kBS * kD * 2);
  _Float16* vfp    = (_Float16*)carve((size_t)kBS * kD * 2);
  _Float16* att    = (_Float16*)carve((size_t)kBS * kD * 2);
  _Float16* hff    = (_Float16*)carve((size_t)kBS * kF * 2);
  if (off > ws_size) return;  // not enough scratch; bail deterministically

  // transpose-convert: src [batch][K][N] f32 -> dst [batch][N][K] f16
  auto convT = [&](const float* src, _Float16* dst, int K, int N, int batch) {
    dim3 g((unsigned)(K / 32), (unsigned)(N / 32), (unsigned)batch);
    convert_transpose_kernel<<<g, 256, 0, stream>>>(src, dst, K, N);
  };
  convT(inp_w1, w_inp1, kKIN, kHID, 1);
  convT(inp_w2, w_inp2, kHID, kD, 1);
  convT(Wq,   w_q,  kD, kD, kL);
  convT(Wk,   w_k,  kD, kD, kL);
  convT(Wv,   w_v,  kD, kD, kL);
  convT(Wo,   w_o,  kD, kD, kL);
  convT(ffw1, w_f1, kD, kF, kL);
  convT(ffw2, w_f2, kF, kD, kL);

  auto gemm = [&](const _Float16* A, const _Float16* BT, const float* bias,
                  const float* resid, float* oF, _Float16* oH,
                  int Md, int Nd, int Kd, int act) {
    dim3 g((unsigned)(Nd / TILE_N), (unsigned)((Md + TILE_M - 1) / TILE_M));
    gemm_f16_wmma_kernel<<<g, 256, 0, stream>>>(A, BT, bias, resid, oF, oH, Md, Nd, Kd, act);
  };

  // -------- input MLP --------
  build_tok_in_kernel<<<(kB * kKIN + 255) / 256, 256, 0, stream>>>(stoch, action, tok_in);
  gemm(tok_in, w_inp1, inp_b1, nullptr, h1, nullptr, kB, kHID, kKIN, 0);
  rmsnorm_kernel<<<kB, 256, 0, stream>>>(h1, inp_nw, h1n, nullptr, kHID, 1 /*silu*/);
  gemm(h1n, w_inp2, inp_b2, nullptr, tok, nullptr, kB, kD, kHID, 0);
  build_seq_kernel<<<(kBS * kD + 255) / 256, 256, 0, stream>>>(memory, tok, seq);

  // -------- transformer layers --------
  for (int l = 0; l < kL; ++l) {
    size_t wO = (size_t)l * kD * kD;
    rmsnorm_kernel<<<kBS, 256, 0, stream>>>(seq, n1w + (size_t)l * kD, xn, nullptr, kD, 0);
    gemm(xn, w_q + wO, bq + (size_t)l * kD, nullptr, nullptr, qf,  kBS, kD, kD, 0);
    gemm(xn, w_k + wO, bk + (size_t)l * kD, nullptr, nullptr, kfp, kBS, kD, kD, 0);
    gemm(xn, w_v + wO, bv + (size_t)l * kD, nullptr, nullptr, vfp, kBS, kD, kD, 0);
    attention_kernel<<<kB * kH, 256, 0, stream>>>(
        qf, kfp, vfp, rel_emb + (size_t)l * (2 * kRM + 1) * kH, att);
    gemm(att, w_o + wO, bo + (size_t)l * kD, seq, seq, nullptr, kBS, kD, kD, 0);
    rmsnorm_kernel<<<kBS, 256, 0, stream>>>(seq, n2w + (size_t)l * kD, xn, nullptr, kD, 0);
    gemm(xn, w_f1 + (size_t)l * kD * kF, ffb1 + (size_t)l * kF,
         nullptr, nullptr, hff, kBS, kF, kD, 2 /*gelu*/);
    gemm(hff, w_f2 + (size_t)l * kF * kD, ffb2 + (size_t)l * kD,
         seq, seq, nullptr, kBS, kD, kF, 0);
  }

  // -------- outputs --------
  final_deter_kernel<<<kB, 256, 0, stream>>>(seq, deter, fnw, out);
  copy_mem_kernel<<<(kB * kMEM * kD + 255) / 256, 256, 0, stream>>>(seq, out + (size_t)kB * kD);
}
